// get_model_22686017257506
// MI455X (gfx1250) — compile-verified, matrix operand order checked by reading, not run on hardware
//
#include <hip/hip_runtime.h>
#include <hip/hip_bf16.h>
#include <stdint.h>

#define KNN 25
// 1/sqrt(1 + 1e-5): fold BN into a per-channel scale
#define BN_RSQRT 0.99999500003749968f

typedef unsigned int u32;
typedef __attribute__((ext_vector_type(16))) __bf16 v16bf;
typedef __attribute__((ext_vector_type(8)))  float  v8f;
typedef __attribute__((ext_vector_type(4)))  u32    v4u;

union Frag { v16bf h; u32 u[8]; v4u q[2]; };

__device__ __forceinline__ u32 pack2bf(float lo, float hi) {
  union { __bf16 h[2]; u32 u; } p;
  p.h[0] = (__bf16)lo;   // bits [15:0]  -> even K
  p.h[1] = (__bf16)hi;   // bits [31:16] -> odd K
  return p.u;
}

__device__ __forceinline__ u32 lds_off(const void* p) {
  // LDS aperture: flat addr[31:0] is the wave-relative LDS byte offset.
  return (u32)(uintptr_t)p;
}

// ---------------------------------------------------------------------------
// bf16 packing kernels. K dimension is padded to Kd dword-pairs (Kd multiple
// of 16 <=> K padded to multiple of 32); padding is zero-filled so the GEMM
// needs no K-tail handling at all.
// ---------------------------------------------------------------------------
__global__ void pack_x_kernel(const float* __restrict__ in, u32* __restrict__ out,
                              int K, int N, int Kd, long total)
{
  long gid = (long)blockIdx.x * blockDim.x + threadIdx.x;
  if (gid >= total) return;
  int n  = (int)(gid % N);
  int kp = (int)((gid / N) % Kd);
  int b  = (int)(gid / ((long)N * Kd));
  int k0 = 2 * kp, k1 = 2 * kp + 1;
  const float* src = in + (size_t)b * K * N;
  float f0 = (k0 < K) ? src[(size_t)k0 * N + n] : 0.f;
  float f1 = (k1 < K) ? src[(size_t)k1 * N + n] : 0.f;
  out[gid] = pack2bf(f0, f1);
}

__global__ void pack_w_kernel(const float* __restrict__ in, u32* __restrict__ out,
                              int M, int K, int Kd, long total)
{
  long gid = (long)blockIdx.x * blockDim.x + threadIdx.x;
  if (gid >= total) return;
  int kp = (int)(gid % Kd);
  int m  = (int)(gid / Kd);
  int k0 = 2 * kp, k1 = k0 + 1;
  const float* row = in + (size_t)m * K;
  float f0 = (k0 < K) ? row[k0] : 0.f;
  float f1 = (k1 < K) ? row[k1] : 0.f;
  out[gid] = pack2bf(f0, f1);
}

// ---------------------------------------------------------------------------
// Shared epilogue: MODE 0 conv1d(+bias), MODE 1 cbr(relu), MODE 2 cbl(leaky).
// ---------------------------------------------------------------------------
template<int MODE>
__device__ __forceinline__ void wmma_epilogue(const v8f& acc, int n, int row0, int half,
                                              int M, int N, float* __restrict__ Y,
                                              const float* __restrict__ convb,
                                              const float* __restrict__ gamma,
                                              const float* __restrict__ beta)
{
  if (n >= N) return;
#pragma unroll
  for (int r = 0; r < 8; ++r) {
    int m = row0 + r + 8 * half;
    if (m >= M) continue;
    float v = acc[r];
    if (MODE == 0) v += convb[m];
    if (MODE == 1) v = (v + convb[m]) * (gamma[m] * BN_RSQRT) + beta[m];
    if (MODE == 2) v = v * (gamma[m] * BN_RSQRT) + beta[m];
    if (MODE == 1) v = fmaxf(v, 0.f);
    if (MODE == 2) v = (v > 0.f) ? v : 0.2f * v;
    Y[(size_t)m * N + n] = v;
  }
}

// ---------------------------------------------------------------------------
// Direct-load WMMA GEMM (small K): wave computes 16x64 strip, 4 wmma / K-step.
// GVS addressing: uniform bases + per-lane 32-bit offsets.
// ---------------------------------------------------------------------------
template<int MODE>
__global__ __launch_bounds__(128) void gemm_wmma_kernel(
    const u32* __restrict__ Wp, const u32* __restrict__ Xp, float* __restrict__ Y,
    int M, int Kd, int N,
    const float* __restrict__ convb, const float* __restrict__ gamma,
    const float* __restrict__ beta)
{
  const int bz = blockIdx.z;
  const u32* Xb = Xp + (size_t)bz * Kd * N;
  Y += (size_t)bz * M * N;
  const int wave = threadIdx.x >> 5;
  const int lane = threadIdx.x & 31;
  const int half = lane >> 4;
  const int l15  = lane & 15;
  const int row0 = blockIdx.y * 16;
  const int colw = blockIdx.x * 256 + wave * 64;

  const int aoff = min(row0 + l15, M - 1) * Kd + 4 * half;
  const int hoff = 4 * half * N;
  const int cc0 = min(colw +  0 + l15, N - 1) + hoff;
  const int cc1 = min(colw + 16 + l15, N - 1) + hoff;
  const int cc2 = min(colw + 32 + l15, N - 1) + hoff;
  const int cc3 = min(colw + 48 + l15, N - 1) + hoff;

  v8f acc0 = {}, acc1 = {}, acc2 = {}, acc3 = {};

  for (int kd = 0; kd < Kd; kd += 16) {
    if (kd + 16 < Kd)
      __builtin_prefetch(Xb + (size_t)(kd + 16) * N + cc0, 0, 1);
    Frag A, B0, B1, B2, B3;
    A.q[0] = *(const v4u*)(Wp + aoff + kd);
    A.q[1] = *(const v4u*)(Wp + aoff + kd + 8);
#pragma unroll
    for (int v = 0; v < 8; ++v) {
      unsigned ro = (unsigned)(kd + ((v < 4) ? v : 8 + (v - 4))) * (unsigned)N;
      B0.u[v] = Xb[ro + cc0];
      B1.u[v] = Xb[ro + cc1];
      B2.u[v] = Xb[ro + cc2];
      B3.u[v] = Xb[ro + cc3];
    }
    acc0 = __builtin_amdgcn_wmma_f32_16x16x32_bf16(false, A.h, false, B0.h, (short)0, acc0, false, false);
    acc1 = __builtin_amdgcn_wmma_f32_16x16x32_bf16(false, A.h, false, B1.h, (short)0, acc1, false, false);
    acc2 = __builtin_amdgcn_wmma_f32_16x16x32_bf16(false, A.h, false, B2.h, (short)0, acc2, false, false);
    acc3 = __builtin_amdgcn_wmma_f32_16x16x32_bf16(false, A.h, false, B3.h, (short)0, acc3, false, false);
  }

  wmma_epilogue<MODE>(acc0, colw +  0 + l15, row0, half, M, N, Y, convb, gamma, beta);
  wmma_epilogue<MODE>(acc1, colw + 16 + l15, row0, half, M, N, Y, convb, gamma, beta);
  wmma_epilogue<MODE>(acc2, colw + 32 + l15, row0, half, M, N, Y, convb, gamma, beta);
  wmma_epilogue<MODE>(acc3, colw + 48 + l15, row0, half, M, N, Y, convb, gamma, beta);
}

// ---------------------------------------------------------------------------
// Async-to-LDS double-buffered WMMA GEMM (large K). Per K-step the block
// stages B (16x256 dwords) + A (16x16 dwords) into LDS with the saddr form of
// global_load_async_to_lds_b128/b64 (uniform SGPR base + per-lane 32-bit byte
// offset -> no 64-bit vector address math), overlapped with compute on the
// other buffer. s_wait_asynccnt 9 retires exactly the current buffer (async
// loads complete in order; 9 ops per wave per stage). Last K-step is peeled
// so the steady-state loop has no conditional waits. B rows padded to 260
// dwords so the two half-wave K-phases hit disjoint LDS banks.
// ---------------------------------------------------------------------------
#define LDSB 260
template<int MODE>
__global__ __launch_bounds__(128) void gemm_wmma_lds_kernel(
    const u32* __restrict__ Wp, const u32* __restrict__ Xp, float* __restrict__ Y,
    int M, int Kd, int N,
    const float* __restrict__ convb, const float* __restrict__ gamma,
    const float* __restrict__ beta)
{
  __shared__ u32 sB[2][16 * LDSB];
  __shared__ u32 sA[2][16 * 16];
  const int bz = blockIdx.z;
  const u32* Xb = Xp + (size_t)bz * Kd * N;
  Y += (size_t)bz * M * N;
  const int tid  = threadIdx.x;
  const int wave = tid >> 5;
  const int lane = tid & 31;
  const int half = lane >> 4;
  const int l15  = lane & 15;
  const int row0 = blockIdx.y * 16;
  const int colb = blockIdx.x * 256;

  // Staging coordinates. B: thread moves 4 dwords x 8 row-pairs; A: 2 dwords.
  const int rB  = tid >> 6;                    // 0/1: row parity within a pair
  const int cB  = (tid & 63) * 4;              // dword col within 256-wide tile
  const u32 voB = (u32)(rB * N + min(colb + cB, N - 4)) * 4u;  // per-lane bytes
  const int rA  = tid >> 3;                    // A row 0..15
  const int cA  = (tid * 2) & 15;              // A dword col (even)
  const u32 voA = (u32)(min(row0 + rA, M - 1) * Kd + cA) * 4u; // per-lane bytes

  auto stage = [&](int buf, int kd) {
#pragma unroll
    for (int j = 0; j < 8; ++j) {
      const u32* base = Xb + (size_t)(kd + 2 * j) * N;          // uniform SGPR
      u32 lo = lds_off(&sB[buf][(2 * j + rB) * LDSB + cB]);
      asm volatile("global_load_async_to_lds_b128 %0, %1, %2"
                   :: "v"(lo), "v"(voB), "s"(base) : "memory");
    }
    const u32* baseA = Wp + kd;                                 // uniform SGPR
    u32 lo = lds_off(&sA[buf][rA * 16 + cA]);
    asm volatile("global_load_async_to_lds_b64 %0, %1, %2"
                 :: "v"(lo), "v"(voA), "s"(baseA) : "memory");
  };

  v8f acc0 = {}, acc1 = {}, acc2 = {}, acc3 = {};

  auto compute = [&](int buf) {
    Frag A, B0, B1, B2, B3;
    const u32* a0 = &sA[buf][l15 * 16 + 4 * half];
    A.q[0] = *(const v4u*)a0;
    A.q[1] = *(const v4u*)(a0 + 8);
    const int cl = wave * 64 + l15;
#pragma unroll
    for (int v = 0; v < 8; ++v) {
      int kvd = ((v < 4) ? v : 8 + (v - 4)) + 4 * half;
      const u32* br = &sB[buf][kvd * LDSB + cl];
      B0.u[v] = br[0];
      B1.u[v] = br[16];
      B2.u[v] = br[32];
      B3.u[v] = br[48];
    }
    acc0 = __builtin_amdgcn_wmma_f32_16x16x32_bf16(false, A.h, false, B0.h, (short)0, acc0, false, false);
    acc1 = __builtin_amdgcn_wmma_f32_16x16x32_bf16(false, A.h, false, B1.h, (short)0, acc1, false, false);
    acc2 = __builtin_amdgcn_wmma_f32_16x16x32_bf16(false, A.h, false, B2.h, (short)0, acc2, false, false);
    acc3 = __builtin_amdgcn_wmma_f32_16x16x32_bf16(false, A.h, false, B3.h, (short)0, acc3, false, false);
  };

  stage(0, 0);
  int kd = 0;
  for (; kd + 16 < Kd; kd += 16) {                 // steady state: no branches
    const int buf = (kd >> 4) & 1;
    stage(buf ^ 1, kd + 16);
    asm volatile("s_wait_asynccnt 0x9" ::: "memory");   // current buffer done
    __syncthreads();
    compute(buf);
    __syncthreads();                               // reads done before restage
  }
  {                                                // peeled last K-step
    const int buf = (kd >> 4) & 1;
    asm volatile("s_wait_asynccnt 0x0" ::: "memory");
    __syncthreads();
    compute(buf);
  }

  const int colw = colb + wave * 64;
  wmma_epilogue<MODE>(acc0, colw +  0 + l15, row0, half, M, N, Y, convb, gamma, beta);
  wmma_epilogue<MODE>(acc1, colw + 16 + l15, row0, half, M, N, Y, convb, gamma, beta);
  wmma_epilogue<MODE>(acc2, colw + 32 + l15, row0, half, M, N, Y, convb, gamma, beta);
  wmma_epilogue<MODE>(acc3, colw + 48 + l15, row0, half, M, N, Y, convb, gamma, beta);
}

// ---------------------------------------------------------------------------
// Farthest point sampling: one block per batch, distance array lives in LDS.
// ---------------------------------------------------------------------------
__global__ __launch_bounds__(256) void fps_kernel(const float* __restrict__ xyz,
                                                  int N, int npoint, int* __restrict__ out)
{
  __shared__ float dist[8192];
  __shared__ float rv[256];
  __shared__ int   ri[256];
  const int b = blockIdx.x;
  const int t = threadIdx.x;
  const float* px = xyz + (size_t)b * 3 * N;
  for (int i = t; i < N; i += 256) dist[i] = 1e10f;
  __syncthreads();
  int far = 0;
  for (int it = 0; it < npoint; ++it) {
    if (t == 0) out[b * npoint + it] = far;
    float cx = px[far], cy = px[N + far], cz = px[2 * N + far];
    float best = -1.f; int bi = 0;
    for (int i = t; i < N; i += 256) {
      float dx = px[i] - cx, dy = px[N + i] - cy, dz = px[2 * N + i] - cz;
      float d = fminf(dist[i], dx * dx + dy * dy + dz * dz);
      dist[i] = d;
      if (d > best) { best = d; bi = i; }
    }
    rv[t] = best; ri[t] = bi;
    __syncthreads();
    for (int s = 128; s > 0; s >>= 1) {
      if (t < s && (rv[t + s] > rv[t] || (rv[t + s] == rv[t] && ri[t + s] < ri[t]))) {
        rv[t] = rv[t + s]; ri[t] = ri[t + s];
      }
      __syncthreads();
    }
    far = ri[0];
    __syncthreads();
  }
}

__global__ void ball_query_kernel(const float* __restrict__ xyz, const float* __restrict__ nxyz,
                                  int N, int S, float r2, int K, int* __restrict__ gidx, int B)
{
  int gid = blockIdx.x * blockDim.x + threadIdx.x;
  if (gid >= B * S) return;
  int b = gid / S, s = gid - b * S;
  const float* px = xyz + (size_t)b * 3 * N;
  float qx = nxyz[((size_t)b * 3 + 0) * S + s];
  float qy = nxyz[((size_t)b * 3 + 1) * S + s];
  float qz = nxyz[((size_t)b * 3 + 2) * S + s];
  int* o = gidx + (size_t)gid * K;
  int cnt = 0, first = 0;
  for (int n = 0; n < N && cnt < K; ++n) {
    float dx = px[n] - qx, dy = px[N + n] - qy, dz = px[2 * N + n] - qz;
    if (dx * dx + dy * dy + dz * dz <= r2) {
      if (cnt == 0) first = n;
      o[cnt++] = n;
    }
  }
  for (; cnt < K; ++cnt) o[cnt] = first;
}

__global__ void group_points_kernel(const float* __restrict__ pts, int C,
                                    const float* __restrict__ xyz, const float* __restrict__ nxyz,
                                    const int* __restrict__ gidx, float* __restrict__ out,
                                    int N, int S, int K, long total)
{
  long gid = (long)blockIdx.x * blockDim.x + threadIdx.x;
  if (gid >= total) return;
  int Cg = C + 3;
  int s = (int)(gid % S);
  int k = (int)((gid / S) % K);
  int c = (int)((gid / ((long)S * K)) % Cg);
  int b = (int)(gid / ((long)S * K * Cg));
  int id = gidx[((size_t)b * S + s) * K + k];
  float v;
  if (c < C) v = pts[((size_t)b * C + c) * N + id];
  else {
    int cc = c - C;
    v = xyz[((size_t)b * 3 + cc) * N + id] - nxyz[((size_t)b * 3 + cc) * S + s];
  }
  out[gid] = v;
}

__global__ void maxreduce_kernel(const float* __restrict__ in, float* __restrict__ out,
                                 int B, int C, int R, int I, int rstride, int istride,
                                 int Ctot, int c0)
{
  long gid = (long)blockIdx.x * blockDim.x + threadIdx.x;
  if (gid >= (long)B * C * I) return;
  int i = (int)(gid % I);
  int c = (int)((gid / I) % C);
  int b = (int)(gid / ((long)I * C));
  const float* p = in + (size_t)(b * C + c) * R * I;
  float m = -3.4e38f;
  for (int r = 0; r < R; ++r) m = fmaxf(m, p[(size_t)r * rstride + (size_t)i * istride]);
  out[((size_t)b * Ctot + c0 + c) * I + i] = m;
}

__global__ __launch_bounds__(128) void knn_kernel(const float* __restrict__ x, int C, int N,
                                                  int k, int* __restrict__ idx)
{
  __shared__ float d[1024];
  __shared__ float rv[128];
  __shared__ int   ri[128];
  const int b = blockIdx.y, i = blockIdx.x, t = threadIdx.x;
  const float* px = x + (size_t)b * C * N;
  for (int j = t; j < N; j += 128) {
    float s = 0.f;
    for (int c = 0; c < C; ++c) {
      float diff = px[(size_t)c * N + i] - px[(size_t)c * N + j];
      s += diff * diff;
    }
    d[j] = s;
  }
  __syncthreads();
  int* o = idx + ((size_t)b * N + i) * k;
  for (int kk = 0; kk < k; ++kk) {
    float best = 3.4e38f; int bi = 0;
    for (int j = t; j < N; j += 128)
      if (d[j] < best) { best = d[j]; bi = j; }
    rv[t] = best; ri[t] = bi;
    __syncthreads();
    for (int s = 64; s > 0; s >>= 1) {
      if (t < s && (rv[t + s] < rv[t] || (rv[t + s] == rv[t] && ri[t + s] < ri[t]))) {
        rv[t] = rv[t + s]; ri[t] = ri[t + s];
      }
      __syncthreads();
    }
    if (t == 0) { o[kk] = ri[0]; d[ri[0]] = 3.4e38f; }
    __syncthreads();
  }
}

__global__ void graph_feature_kernel(const float* __restrict__ x, const int* __restrict__ idx,
                                     float* __restrict__ out, int C, int N, int k, long total)
{
  long gid = (long)blockIdx.x * blockDim.x + threadIdx.x;
  if (gid >= total) return;
  int kk = (int)(gid % k);
  int n  = (int)((gid / k) % N);
  int c  = (int)((gid / ((long)k * N)) % (2 * C));
  int b  = (int)(gid / ((long)k * N * 2 * C));
  float v;
  if (c < C) {
    int j = idx[((size_t)b * N + n) * k + kk];
    v = x[((size_t)b * C + c) * N + j] - x[((size_t)b * C + c) * N + n];
  } else {
    v = x[((size_t)b * C + (c - C)) * N + n];
  }
  out[gid] = v;
}

__global__ void three_nn_kernel(const float* __restrict__ x1, const float* __restrict__ x2,
                                int N1, int N2, int* __restrict__ idx, float* __restrict__ w, int B)
{
  int gid = blockIdx.x * blockDim.x + threadIdx.x;
  if (gid >= B * N1) return;
  int b = gid / N1, i = gid - b * N1;
  float ax = x1[((size_t)b * 3 + 0) * N1 + i];
  float ay = x1[((size_t)b * 3 + 1) * N1 + i];
  float az = x1[((size_t)b * 3 + 2) * N1 + i];
  const float* px = x2 + (size_t)b * 3 * N2;
  float d0 = 3.4e38f, d1 = 3.4e38f, d2 = 3.4e38f;
  int i0 = 0, i1 = 0, i2 = 0;
  for (int j = 0; j < N2; ++j) {
    float dx = px[j] - ax, dy = px[N2 + j] - ay, dz = px[2 * N2 + j] - az;
    float d = dx * dx + dy * dy + dz * dz;
    if (d < d0)      { d2 = d1; i2 = i1; d1 = d0; i1 = i0; d0 = d; i0 = j; }
    else if (d < d1) { d2 = d1; i2 = i1; d1 = d;  i1 = j; }
    else if (d < d2) { d2 = d;  i2 = j; }
  }
  float w0 = 1.f / (d0 + 1e-8f), w1 = 1.f / (d1 + 1e-8f), w2 = 1.f / (d2 + 1e-8f);
  float ws = w0 + w1 + w2;
  size_t o = (size_t)gid * 3;
  idx[o] = i0; idx[o + 1] = i1; idx[o + 2] = i2;
  w[o] = w0 / ws; w[o + 1] = w1 / ws; w[o + 2] = w2 / ws;
}

__global__ void interp_concat_kernel(const float* __restrict__ p1, int C1,
                                     const float* __restrict__ p2, int C2,
                                     const int* __restrict__ idx, const float* __restrict__ w,
                                     float* __restrict__ out, int N1, int N2, long total)
{
  long gid = (long)blockIdx.x * blockDim.x + threadIdx.x;
  if (gid >= total) return;
  int n = (int)(gid % N1);
  int c = (int)((gid / N1) % (C1 + C2));
  int b = (int)(gid / ((long)N1 * (C1 + C2)));
  float v;
  if (c < C1) v = p1[((size_t)b * C1 + c) * N1 + n];
  else {
    const float* row = p2 + ((size_t)b * C2 + (c - C1)) * N2;
    size_t o = ((size_t)b * N1 + n) * 3;
    v = row[idx[o]] * w[o] + row[idx[o + 1]] * w[o + 1] + row[idx[o + 2]] * w[o + 2];
  }
  out[gid] = v;
}

__global__ void gather_points_kernel(const float* __restrict__ src, const int* __restrict__ idx,
                                     float* __restrict__ dst, int C, int N, int S, int B)
{
  long gid = (long)blockIdx.x * blockDim.x + threadIdx.x;
  if (gid >= (long)B * C * S) return;
  int s = (int)(gid % S);
  int c = (int)((gid / S) % C);
  int b = (int)(gid / ((long)S * C));
  dst[gid] = src[((size_t)b * C + c) * N + idx[(size_t)b * S + s]];
}

__global__ void slice_channels_kernel(const float* __restrict__ src, float* __restrict__ dst,
                                      int Csrc, int c0, int C, int N, int B)
{
  long gid = (long)blockIdx.x * blockDim.x + threadIdx.x;
  if (gid >= (long)B * C * N) return;
  int n = (int)(gid % N);
  int c = (int)((gid / N) % C);
  int b = (int)(gid / ((long)N * C));
  dst[gid] = src[((size_t)b * Csrc + c0 + c) * N + n];
}

// ---------------------------------------------------------------------------
// Host side
// ---------------------------------------------------------------------------
struct CBR  { const float *b, *be, *g, *w; };
struct CBNB { const float *be, *g, *w; };
struct CONVP{ const float *b, *w; };

extern "C" void kernel_launch(void* const* d_in, const int* in_sizes, int n_in,
                              void* d_out, int out_size, void* d_ws, size_t ws_size,
                              hipStream_t stream)
{
  (void)in_sizes; (void)n_in; (void)out_size; (void)ws_size;
  const int B = 2, N0 = 8192;
  const float* xyz = (const float*)d_in[0];   // (B, 6, N0)

  int pi = 1;
  auto F = [&]() { return (const float*)d_in[pi++]; };
  auto takeCBR  = [&]() { CBR  p; p.b = F(); p.be = F(); p.g = F(); p.w = F(); return p; };
  auto takeCBNB = [&]() { CBNB p; p.be = F(); p.g = F(); p.w = F(); return p; };
  auto takeCONV = [&]() { CONVP p; p.b = F(); p.w = F(); return p; };

  CBR   cls1 = takeCBR();   CONVP cls2 = takeCONV();
  CBNB  conv3 = takeCBNB(), conv4 = takeCBNB(), conv5 = takeCBNB(),
        conv6 = takeCBNB(), conv7 = takeCBNB();
  CBR   dist1 = takeCBR();  CONVP dist2 = takeCONV();
  CBR   fp1a = takeCBR(), fp1b = takeCBR();
  CBR   fp2a = takeCBR(), fp2b = takeCBR();
  CBR   fp3a = takeCBR(), fp3b = takeCBR();
  CBR   off1 = takeCBR();   CONVP off2 = takeCONV();
  CBR   sa1_00 = takeCBR(), sa1_01 = takeCBR(), sa1_10 = takeCBR(), sa1_11 = takeCBR();
  CBR   sa2_00 = takeCBR(), sa2_01 = takeCBR(), sa2_10 = takeCBR(), sa2_11 = takeCBR();
  CBR   sa3_00 = takeCBR(), sa3_01 = takeCBR(), sa3_10 = takeCBR(), sa3_11 = takeCBR();

  float* out_l0p = (float*)d_out;
  float* out_l3p = out_l0p + (size_t)B * 128 * N0;
  float* out_l0x = out_l3p + (size_t)B * 2048 * 256;
  float* out_l3x = out_l0x + (size_t)B * 3 * N0;
  float* out_off = out_l3x + (size_t)B * 3 * 256;
  float* out_dst = out_off + (size_t)B * 3 * N0;
  float* out_cls = out_dst + (size_t)B * 1 * N0;

  size_t ws_off = 0;
  auto allocf = [&](size_t n) {
    float* p = (float*)((char*)d_ws + ws_off);
    ws_off += ((n * sizeof(float)) + 255) & ~(size_t)255;
    return p;
  };
  auto alloci = [&](size_t n) {
    int* p = (int*)((char*)d_ws + ws_off);
    ws_off += ((n * sizeof(int)) + 255) & ~(size_t)255;
    return p;
  };

  float* l1_xyz = allocf((size_t)B * 3 * 1024);
  float* l2_xyz = allocf((size_t)B * 3 * 512);
  float* l1p_sa = allocf((size_t)B * 64 * 1024);
  float* l1p    = allocf((size_t)B * 64 * 1024);
  float* l2p_sa = allocf((size_t)B * 256 * 512);
  float* l2p    = allocf((size_t)B * 256 * 512);
  float* l3p_sa = allocf((size_t)B * 1024 * 256);
  float* l2p_fp = allocf((size_t)B * 1024 * 512);
  float* l1p_fp = allocf((size_t)B * 512 * 1024);
  float* htmp   = allocf((size_t)B * 17 * N0);
  int* fidx1 = alloci((size_t)B * 1024);
  int* fidx2 = alloci((size_t)B * 512);
  int* fidx3 = alloci((size_t)B * 256);
  const size_t mark = ws_off;

  auto grid1 = [](long t) { return dim3((unsigned)((t + 255) / 256)); };

  auto gemm = [&](const float* W, const float* X, float* Y, int M, int K, int N,
                  const float* cb, const float* g, const float* be, int mode) {
    const int Kd = ((K + 31) & ~31) / 2;
    u32* Wp = (u32*)allocf((size_t)M * Kd);
    u32* Xp = (u32*)allocf((size_t)B * Kd * N);
    long tw = (long)M * Kd;
    pack_w_kernel<<<grid1(tw), 256, 0, stream>>>(W, Wp, M, K, Kd, tw);
    long tx = (long)B * Kd * N;
    pack_x_kernel<<<grid1(tx), 256, 0, stream>>>(X, Xp, K, N, Kd, tx);
    dim3 grid((unsigned)((N + 255) / 256), (unsigned)((M + 15) / 16), (unsigned)B);
    dim3 blk(128, 1, 1);
    if (Kd >= 32) {   // large K: async-to-LDS double-buffered pipeline
      if (mode == 0)      gemm_wmma_lds_kernel<0><<<grid, blk, 0, stream>>>(Wp, Xp, Y, M, Kd, N, cb, g, be);
      else if (mode == 1) gemm_wmma_lds_kernel<1><<<grid, blk, 0, stream>>>(Wp, Xp, Y, M, Kd, N, cb, g, be);
      else                gemm_wmma_lds_kernel<2><<<grid, blk, 0, stream>>>(Wp, Xp, Y, M, Kd, N, cb, g, be);
    } else {
      if (mode == 0)      gemm_wmma_kernel<0><<<grid, blk, 0, stream>>>(Wp, Xp, Y, M, Kd, N, cb, g, be);
      else if (mode == 1) gemm_wmma_kernel<1><<<grid, blk, 0, stream>>>(Wp, Xp, Y, M, Kd, N, cb, g, be);
      else                gemm_wmma_kernel<2><<<grid, blk, 0, stream>>>(Wp, Xp, Y, M, Kd, N, cb, g, be);
    }
  };

  auto sa_branch = [&](const float* xz, const float* pts, int C, int N,
                       const float* nxz, int S, float radius, int K,
                       const CBR& c0p, const CBR& c1p, int M0, int M1,
                       float* outP, int Ctot, int coff) {
    int* gidx = alloci((size_t)B * S * K);
    ball_query_kernel<<<grid1((long)B * S), 256, 0, stream>>>(
        xz, nxz, N, S, radius * radius, K, gidx, B);
    int Cg = C + 3;
    long tg = (long)B * Cg * K * S;
    float* gbuf = allocf((size_t)tg);
    group_points_kernel<<<grid1(tg), 256, 0, stream>>>(pts, C, xz, nxz, gidx, gbuf, N, S, K, tg);
    float* gm = allocf((size_t)B * M0 * K * S);
    gemm(c0p.w, gbuf, gm, M0, Cg, K * S, c0p.b, c0p.g, c0p.be, 1);
    float* go = allocf((size_t)B * M1 * K * S);
    gemm(c1p.w, gm, go, M1, M0, K * S, c1p.b, c1p.g, c1p.be, 1);
    long tm = (long)B * M1 * S;
    maxreduce_kernel<<<grid1(tm), 256, 0, stream>>>(go, outP, B, M1, K, S, S, 1, Ctot, coff);
  };

  auto graph_block = [&](const float* x, int C, int N, const CBNB* convs, const int* Ms,
                         int nconv, float* outP) {
    int* idx = alloci((size_t)B * N * KNN);
    knn_kernel<<<dim3((unsigned)N, (unsigned)B), 128, 0, stream>>>(x, C, N, KNN, idx);
    long tg = (long)B * 2 * C * N * KNN;
    float* gf = allocf((size_t)tg);
    graph_feature_kernel<<<grid1(tg), 256, 0, stream>>>(x, idx, gf, C, N, KNN, tg);
    const float* cur = gf;
    int Kc = 2 * C;
    for (int i = 0; i < nconv; ++i) {
      float* o = allocf((size_t)B * Ms[i] * N * KNN);
      gemm(convs[i].w, cur, o, Ms[i], Kc, N * KNN, nullptr, convs[i].g, convs[i].be, 2);
      cur = o; Kc = Ms[i];
    }
    long tm = (long)B * Kc * N;
    maxreduce_kernel<<<grid1(tm), 256, 0, stream>>>(cur, outP, B, Kc, KNN, N, 1, KNN, Kc, 0);
  };

  auto fp_block = [&](const float* x1, int N1, const float* x2, int N2,
                      const float* p1, int C1, const float* p2, int C2,
                      const CBR& ca, const CBR& cbp, int Ma, int Mb, float* outP) {
    int* idx = alloci((size_t)B * N1 * 3);
    float* w = allocf((size_t)B * N1 * 3);
    three_nn_kernel<<<grid1((long)B * N1), 256, 0, stream>>>(x1, x2, N1, N2, idx, w, B);
    long tc = (long)B * (C1 + C2) * N1;
    float* cat = allocf((size_t)tc);
    interp_concat_kernel<<<grid1(tc), 256, 0, stream>>>(p1, C1, p2, C2, idx, w, cat, N1, N2, tc);
    float* mid = allocf((size_t)B * Ma * N1);
    gemm(ca.w, cat, mid, Ma, C1 + C2, N1, ca.b, ca.g, ca.be, 1);
    gemm(cbp.w, mid, outP, Mb, Ma, N1, cbp.b, cbp.g, cbp.be, 1);
  };

  slice_channels_kernel<<<grid1((long)B * 3 * N0), 256, 0, stream>>>(xyz, out_l0x, 6, 0, 3, N0, B);

  fps_kernel<<<B, 256, 0, stream>>>(out_l0x, N0, 1024, fidx1);
  gather_points_kernel<<<grid1((long)B * 3 * 1024), 256, 0, stream>>>(out_l0x, fidx1, l1_xyz, 3, N0, 1024, B);
  sa_branch(out_l0x, xyz, 6, N0, l1_xyz, 1024, 0.025f, 32, sa1_00, sa1_01, 16, 32, l1p_sa, 64, 0);
  ws_off = mark;
  sa_branch(out_l0x, xyz, 6, N0, l1_xyz, 1024, 0.05f, 64, sa1_10, sa1_11, 32, 32, l1p_sa, 64, 32);
  ws_off = mark;

  { CBNB cc[2] = {conv3, conv4}; int mm[2] = {64, 64};
    graph_block(l1p_sa, 64, 1024, cc, mm, 2, l1p); ws_off = mark; }

  fps_kernel<<<B, 256, 0, stream>>>(l1_xyz, 1024, 512, fidx2);
  gather_points_kernel<<<grid1((long)B * 3 * 512), 256, 0, stream>>>(l1_xyz, fidx2, l2_xyz, 3, 1024, 512, B);
  sa_branch(l1_xyz, l1p, 64, 1024, l2_xyz, 512, 0.05f, 32, sa2_00, sa2_01, 64, 128, l2p_sa, 256, 0);
  ws_off = mark;
  sa_branch(l1_xyz, l1p, 64, 1024, l2_xyz, 512, 0.1f, 64, sa2_10, sa2_11, 64, 128, l2p_sa, 256, 128);
  ws_off = mark;

  { CBNB cc[2] = {conv5, conv6}; int mm[2] = {256, 256};
    graph_block(l2p_sa, 256, 512, cc, mm, 2, l2p); ws_off = mark; }

  fps_kernel<<<B, 256, 0, stream>>>(l2_xyz, 512, 256, fidx3);
  gather_points_kernel<<<grid1((long)B * 3 * 256), 256, 0, stream>>>(l2_xyz, fidx3, out_l3x, 3, 512, 256, B);
  sa_branch(l2_xyz, l2p, 256, 512, out_l3x, 256, 0.1f, 32, sa3_00, sa3_01, 256, 512, l3p_sa, 1024, 0);
  ws_off = mark;
  sa_branch(l2_xyz, l2p, 256, 512, out_l3x, 256, 0.2f, 64, sa3_10, sa3_11, 256, 512, l3p_sa, 1024, 512);
  ws_off = mark;

  { CBNB cc[1] = {conv7}; int mm[1] = {2048};
    graph_block(l3p_sa, 1024, 256, cc, mm, 1, out_l3p); ws_off = mark; }

  fp_block(l2_xyz, 512, out_l3x, 256, l2p, 256, out_l3p, 2048, fp3a, fp3b, 1024, 1024, l2p_fp);
  ws_off = mark;
  fp_block(l1_xyz, 1024, l2_xyz, 512, l1p, 64, l2p_fp, 1024, fp2a, fp2b, 512, 512, l1p_fp);
  ws_off = mark;
  fp_block(out_l0x, N0, l1_xyz, 1024, xyz, 6, l1p_fp, 512, fp1a, fp1b, 256, 128, out_l0p);
  ws_off = mark;

  gemm(off1.w,  out_l0p, htmp,    16, 128, N0, off1.b,  off1.g,  off1.be,  1);
  gemm(off2.w,  htmp,    out_off,  3,  16, N0, off2.b,  nullptr, nullptr,  0);
  gemm(dist1.w, out_l0p, htmp,    16, 128, N0, dist1.b, dist1.g, dist1.be, 1);
  gemm(dist2.w, htmp,    out_dst,  1,  16, N0, dist2.b, nullptr, nullptr,  0);
  gemm(cls1.w,  out_l0p, htmp,    17, 128, N0, cls1.b,  cls1.g,  cls1.be,  1);
  gemm(cls2.w,  htmp,    out_cls, 17,  17, N0, cls2.b,  nullptr, nullptr,  0);
}
